// HKANGNN_83184926589409
// MI455X (gfx1250) — matrix operand order repaired and checked by reading, not compiled, not run on hardware
//
#include <hip/hip_runtime.h>
#include <hip/hip_bf16.h>
#include <math.h>

typedef __attribute__((ext_vector_type(16))) _Float16 v16h;
typedef __attribute__((ext_vector_type(8)))  _Float16 v8h;
typedef __attribute__((ext_vector_type(4)))  _Float16 v4h;
typedef __attribute__((ext_vector_type(8)))  float    v8f;

#define HDIM 128

// ---------------------------------------------------------------- fragment helpers
// A fragment from an LDS-resident f16 row-major tile (ld halves per row).
// lane<16: M=lane, K = k0+[0..7] ++ k0+[16..23]; lane>=16: same M, K shifted +8.
__device__ __forceinline__ v16h load_a_frag_lds(const _Float16* a_sh, int ld, int lane, int k0) {
    const _Float16* p = a_sh + (lane & 15) * ld + k0 + ((lane & 16) ? 8 : 0);
    v8h h0 = *(const v8h*)(p);
    v8h h1 = *(const v8h*)(p + 16);
    return __builtin_shufflevector(h0, h1, 0,1,2,3,4,5,6,7,8,9,10,11,12,13,14,15);
}

// B fragment from a pre-converted f16 weight matrix W (O x K row-major), B[k][n]=W[n][k].
// lane<16: N=lane, K = k0+[0..7] ++ k0+[16..23]; lane>=16: N=lane-16, K shifted +8.
__device__ __forceinline__ v16h load_b_frag_h(const _Float16* __restrict__ Wh, int ld,
                                              int col, int k0, int lane) {
    const _Float16* p = Wh + (size_t)col * ld + k0 + ((lane & 16) ? 8 : 0);
    v8h h0 = *(const v8h*)(p);
    v8h h1 = *(const v8h*)(p + 16);
    return __builtin_shufflevector(h0, h1, 0,1,2,3,4,5,6,7,8,9,10,11,12,13,14,15);
}

// Cooperative f32 -> f16 tile staging into LDS (nfloat4 = total floats / 4).
__device__ __forceinline__ void stage_tile_f16(const float* __restrict__ src, _Float16* dst,
                                               int nfloat4, int tid, int nthreads) {
    const float4* s = (const float4*)src;
    for (int i = tid; i < nfloat4; i += nthreads) {
        float4 v = s[i];
        v4h h;
        h[0] = (_Float16)v.x; h[1] = (_Float16)v.y;
        h[2] = (_Float16)v.z; h[3] = (_Float16)v.w;
        *(v4h*)(dst + 4 * i) = h;
    }
}

// ---------------------------------------------------------------- B-spline (G=5,K=3 -> 8 bases)
__device__ __forceinline__ void bspline8(float x, float* out) {
    float b[11];
#pragma unroll
    for (int i = 0; i < 11; ++i) {
        float t0 = 0.4f * i - 2.2f;
        b[i] = (x >= t0 && x < t0 + 0.4f) ? 1.0f : 0.0f;
    }
#pragma unroll
    for (int j = 1; j <= 3; ++j) {
        float inv = 1.0f / (0.4f * j);
#pragma unroll
        for (int i = 0; i + j < 11; ++i) {
            float ti  = 0.4f * i - 2.2f;
            float tip = ti + 0.4f * (j + 1);
            b[i] = (x - ti) * inv * b[i] + (tip - x) * inv * b[i + 1];
        }
    }
#pragma unroll
    for (int i = 0; i < 8; ++i) out[i] = b[i];
}

__device__ __forceinline__ float siluf(float x) { return x / (1.0f + expf(-x)); }

// ---------------------------------------------------------------- f32 -> f16 weight conversion
__global__ void cvt_f16_kernel(const float* __restrict__ src, _Float16* __restrict__ dst, int n4) {
    int i = blockIdx.x * blockDim.x + threadIdx.x;
    if (i >= n4) return;
    float4 v = ((const float4*)src)[i];
    v4h h;
    h[0] = (_Float16)v.x; h[1] = (_Float16)v.y;
    h[2] = (_Float16)v.z; h[3] = (_Float16)v.w;
    ((v4h*)dst)[i] = h;
}

__global__ void wsum_f16_kernel(const float* __restrict__ a, const float* __restrict__ b,
                                _Float16* __restrict__ dst, int n4) {
    int i = blockIdx.x * blockDim.x + threadIdx.x;
    if (i >= n4) return;
    float4 va = ((const float4*)a)[i];
    float4 vb = ((const float4*)b)[i];
    v4h h;
    h[0] = (_Float16)(va.x + vb.x); h[1] = (_Float16)(va.y + vb.y);
    h[2] = (_Float16)(va.z + vb.z); h[3] = (_Float16)(va.w + vb.w);
    ((v4h*)dst)[i] = h;
}

// ---------------------------------------------------------------- e = tanh(X @ W^T + b)
// X: (M,768) f32, Wh: (128,768) f16, out: (M,128). grid = M/16, block = 256 (8 waves).
__global__ void email_embed_kernel(const float* __restrict__ X, const _Float16* __restrict__ Wh,
                                   const float* __restrict__ bias, float* __restrict__ out) {
    __shared__ _Float16 a_sh[16 * 768];            // 24 KB
    int tid = threadIdx.x;
    int m0 = blockIdx.x * 16;
    stage_tile_f16(X + (size_t)m0 * 768, a_sh, 16 * 768 / 4, tid, 256);
    __syncthreads();

    int lane = tid & 31;
    int wave = tid >> 5;
    int col = wave * 16 + (lane & 15);
    const _Float16* brow = Wh + (size_t)col * 768;
    v8f c = {};
    for (int k0 = 0; k0 < 768; k0 += 32) {
        if (k0 + 32 < 768) __builtin_prefetch(brow + k0 + 32, 0, 0);
        v16h a = load_a_frag_lds(a_sh, 768, lane, k0);
        v16h b = load_b_frag_h(Wh, 768, col, k0, lane);
        c = __builtin_amdgcn_wmma_f32_16x16x32_f16(false, a, false, b, (short)0, c, false, false);
    }
    float bn = bias[col];
#pragma unroll
    for (int j = 0; j < 8; ++j) {
        int r = m0 + ((lane < 16) ? j : (j + 8));
        out[(size_t)r * HDIM + col] = tanhf(c[j] + bn);
    }
}

// ---------------------------------------------------------------- email_h GEMM (3 fused K=128 segments)
// h = lrelu(0.5*(agg_se@Wl_se^T + agg_ue@Wl_ue^T + e@(Wr_se+Wr_ue)^T + bl_se + bl_ue))
__global__ void email_h_kernel(const float* __restrict__ agg_se, const float* __restrict__ agg_ue,
                               const float* __restrict__ e_in,
                               const _Float16* __restrict__ Wh_se, const _Float16* __restrict__ Wh_ue,
                               const _Float16* __restrict__ Wh_sum,
                               const float* __restrict__ bl_se, const float* __restrict__ bl_ue,
                               float* __restrict__ out) {
    __shared__ _Float16 a_sh[3 * 16 * 128];        // 12 KB
    int tid = threadIdx.x;
    int m0 = blockIdx.x * 16;
    stage_tile_f16(agg_se + (size_t)m0 * HDIM, a_sh + 0 * 2048, 512, tid, 256);
    stage_tile_f16(agg_ue + (size_t)m0 * HDIM, a_sh + 1 * 2048, 512, tid, 256);
    stage_tile_f16(e_in   + (size_t)m0 * HDIM, a_sh + 2 * 2048, 512, tid, 256);
    __syncthreads();

    int lane = tid & 31;
    int wave = tid >> 5;
    int col = wave * 16 + (lane & 15);
    const _Float16* Bs[3] = { Wh_se, Wh_ue, Wh_sum };
    v8f c = {};
#pragma unroll
    for (int s = 0; s < 3; ++s) {
        const _Float16* B = Bs[s];
        const _Float16* a_seg = a_sh + s * 2048;
#pragma unroll
        for (int k0 = 0; k0 < 128; k0 += 32) {
            v16h a = load_a_frag_lds(a_seg, 128, lane, k0);
            v16h b = load_b_frag_h(B, 128, col, k0, lane);
            c = __builtin_amdgcn_wmma_f32_16x16x32_f16(false, a, false, b, (short)0, c, false, false);
        }
    }
    float bn = bl_se[col] + bl_ue[col];
#pragma unroll
    for (int j = 0; j < 8; ++j) {
        int r = m0 + ((lane < 16) ? j : (j + 8));
        float v = 0.5f * (c[j] + bn);
        out[(size_t)r * HDIM + col] = (v >= 0.0f) ? v : 0.2f * v;
    }
}

// ---------------------------------------------------------------- KAN feature layer: tanh(kan(x))
template <int D>
__global__ void kan_tanh_kernel(const float* __restrict__ X, const float* __restrict__ base_w,
                                const float* __restrict__ spline_w, float* __restrict__ out) {
    __shared__ float s_b[D * 8];
    __shared__ float s_s[D];
    int row = blockIdx.x;
    int t = threadIdx.x;
    if (t < D) {
        float x = X[(size_t)row * D + t];
        float bs[8];
        bspline8(x, bs);
#pragma unroll
        for (int b = 0; b < 8; ++b) s_b[t * 8 + b] = bs[b];
        s_s[t] = siluf(x);
    }
    __syncthreads();
    float acc = 0.0f;
#pragma unroll
    for (int d = 0; d < D; ++d) {
        acc += s_s[d] * base_w[t * D + d];
        const float* sw = spline_w + ((size_t)t * D + d) * 8;
        float bacc = 0.0f;
#pragma unroll
        for (int b = 0; b < 8; ++b) bacc += s_b[d * 8 + b] * sw[b];
        acc += bacc;
    }
    out[(size_t)row * HDIM + t] = tanhf(acc);
}

// ---------------------------------------------------------------- edge scatter (mean aggregation)
__global__ void scatter_add_kernel(const float* __restrict__ xsrc, const int* __restrict__ src,
                                   const int* __restrict__ dst, float* __restrict__ acc, long total) {
    long idx = (long)blockIdx.x * blockDim.x + threadIdx.x;
    if (idx >= total) return;
    int e = (int)(idx >> 7);
    int c = (int)(idx & 127);
    atomicAdd(&acc[(size_t)dst[e] * HDIM + c], xsrc[(size_t)src[e] * HDIM + c]);
}

__global__ void count_kernel(const int* __restrict__ dst, float* __restrict__ cnt, int E) {
    int i = blockIdx.x * blockDim.x + threadIdx.x;
    if (i >= E) return;
    atomicAdd(&cnt[dst[i]], 1.0f);
}

__global__ void norm_mean_kernel(float* __restrict__ acc, const float* __restrict__ cnt, long total) {
    long idx = (long)blockIdx.x * blockDim.x + threadIdx.x;
    if (idx >= total) return;
    acc[idx] /= fmaxf(cnt[idx >> 7], 1.0f);
}

// ---------------------------------------------------------------- BN stats over combined = [h | e]
// Row-coalesced: thread t owns combined channel t; rows grid-strided over blocks.
__global__ void bn_partial_kernel(const float* __restrict__ h, const float* __restrict__ e,
                                  float* __restrict__ sums, float* __restrict__ sqs, int N) {
    int t = threadIdx.x;  // 0..255
    float sum = 0.0f, sq = 0.0f;
    for (int r = blockIdx.x; r < N; r += gridDim.x) {
        float v = (t < HDIM) ? h[(size_t)r * HDIM + t] : e[(size_t)r * HDIM + (t - HDIM)];
        sum += v;
        sq += v * v;
    }
    atomicAdd(&sums[t], sum);
    atomicAdd(&sqs[t], sq);
}

__global__ void bn_finalize_kernel(const float* __restrict__ sums, const float* __restrict__ sqs,
                                   float* __restrict__ stats, int N) {
    int c = threadIdx.x;  // 0..255
    float mu = sums[c] / (float)N;
    float var = sqs[c] / (float)N - mu * mu;
    stats[c] = mu;
    stats[256 + c] = rsqrtf(var + 1e-5f);
}

// ---------------------------------------------------------------- fused BN + KAN classifier
__global__ void classify_kernel(const float* __restrict__ h, const float* __restrict__ e,
                                const float* __restrict__ stats,
                                const float* __restrict__ gamma, const float* __restrict__ beta,
                                const float* __restrict__ bw, const float* __restrict__ sw,
                                float* __restrict__ out) {
    __shared__ float r0[256];
    __shared__ float r1[256];
    int row = blockIdx.x;
    int c = threadIdx.x;
    float x = (c < HDIM) ? h[(size_t)row * HDIM + c] : e[(size_t)row * HDIM + (c - HDIM)];
    float xn = (x - stats[c]) * stats[256 + c] * gamma[c] + beta[c];
    float bs[8];
    bspline8(xn, bs);
    float sil = siluf(xn);
    float p0 = sil * bw[c];
    float p1 = sil * bw[256 + c];
    const float* s0 = sw + (size_t)c * 8;
    const float* s1 = sw + (size_t)(256 + c) * 8;
#pragma unroll
    for (int b = 0; b < 8; ++b) { p0 += bs[b] * s0[b]; p1 += bs[b] * s1[b]; }
    r0[c] = p0; r1[c] = p1;
    __syncthreads();
    for (int s = 128; s > 0; s >>= 1) {
        if (c < s) { r0[c] += r0[c + s]; r1[c] += r1[c + s]; }
        __syncthreads();
    }
    if (c == 0) {
        out[(size_t)row * 2 + 0] = r0[0];
        out[(size_t)row * 2 + 1] = r1[0];
    }
}

// ---------------------------------------------------------------- launch
extern "C" void kernel_launch(void* const* d_in, const int* in_sizes, int n_in,
                              void* d_out, int out_size, void* d_ws, size_t ws_size,
                              hipStream_t stream) {
    (void)n_in; (void)out_size; (void)ws_size;
    const float* email_x   = (const float*)d_in[0];
    const float* url_x     = (const float*)d_in[1];
    const float* sender_x  = (const float*)d_in[2];
    const float* W_email   = (const float*)d_in[3];
    const float* b_email   = (const float*)d_in[4];
    const float* url_bw    = (const float*)d_in[5];
    const float* url_sw    = (const float*)d_in[6];
    const float* snd_bw    = (const float*)d_in[7];
    const float* snd_sw    = (const float*)d_in[8];
    const float* Wl_se     = (const float*)d_in[9];
    const float* bl_se     = (const float*)d_in[10];
    const float* Wr_se     = (const float*)d_in[11];
    // d_in[12..14] = Wl_eu/bl_eu/Wr_eu: dead downstream (url_h unused) -> skipped
    const float* Wl_ue     = (const float*)d_in[15];
    const float* bl_ue     = (const float*)d_in[16];
    const float* Wr_ue     = (const float*)d_in[17];
    const float* bn_gamma  = (const float*)d_in[18];
    const float* bn_beta   = (const float*)d_in[19];
    const float* cls_bw    = (const float*)d_in[20];
    const float* cls_sw    = (const float*)d_in[21];
    const int*   se_src    = (const int*)d_in[22];
    const int*   se_dst    = (const int*)d_in[23];
    // d_in[24..25] = eu edges: skipped (dead)
    const int*   ue_src    = (const int*)d_in[26];
    const int*   ue_dst    = (const int*)d_in[27];

    const int N_E  = in_sizes[0] / 768;   // 30000 (multiple of 16)
    const int N_U  = in_sizes[1] / 8;     // 200000
    const int N_S  = in_sizes[2];         // 20000
    const int E_SE = in_sizes[22];        // 30000
    const int E_UE = in_sizes[26];        // 1000000

    float* ws      = (float*)d_ws;
    float* e_buf   = ws;                                 // N_E*128
    float* u_buf   = e_buf  + (size_t)N_E * HDIM;        // N_U*128
    float* s_buf   = u_buf  + (size_t)N_U * HDIM;        // N_S*128
    float* acc_se  = s_buf  + (size_t)N_S * HDIM;        // N_E*128
    float* acc_ue  = acc_se + (size_t)N_E * HDIM;        // N_E*128
    float* h_buf   = acc_ue + (size_t)N_E * HDIM;        // N_E*128
    float* cnt_se  = h_buf  + (size_t)N_E * HDIM;        // N_E
    float* cnt_ue  = cnt_se + N_E;                       // N_E
    float* stats   = cnt_ue + N_E;                       // 512 (mu || rstd)
    float* bn_sums = stats  + 512;                       // 256
    float* bn_sqs  = bn_sums + 256;                      // 256
    _Float16* Wh_email = (_Float16*)(bn_sqs + 256);      // 128*768 halves
    _Float16* Wh_l_se  = Wh_email + HDIM * 768;          // 128*128
    _Float16* Wh_l_ue  = Wh_l_se + HDIM * HDIM;
    _Float16* Wh_sum   = Wh_l_ue + HDIM * HDIM;

    // zero accumulators (graph-capturable memset nodes)
    hipMemsetAsync(acc_se, 0, (size_t)N_E * HDIM * sizeof(float), stream);
    hipMemsetAsync(acc_ue, 0, (size_t)N_E * HDIM * sizeof(float), stream);
    hipMemsetAsync(cnt_se, 0, (size_t)N_E * sizeof(float), stream);
    hipMemsetAsync(cnt_ue, 0, (size_t)N_E * sizeof(float), stream);
    hipMemsetAsync(bn_sums, 0, 512 * sizeof(float), stream);

    // one-time f16 weight conversions
    cvt_f16_kernel<<<(HDIM * 768 / 4 + 255) / 256, 256, 0, stream>>>(W_email, Wh_email, HDIM * 768 / 4);
    cvt_f16_kernel<<<(HDIM * HDIM / 4 + 255) / 256, 256, 0, stream>>>(Wl_se, Wh_l_se, HDIM * HDIM / 4);
    cvt_f16_kernel<<<(HDIM * HDIM / 4 + 255) / 256, 256, 0, stream>>>(Wl_ue, Wh_l_ue, HDIM * HDIM / 4);
    wsum_f16_kernel<<<(HDIM * HDIM / 4 + 255) / 256, 256, 0, stream>>>(Wr_se, Wr_ue, Wh_sum, HDIM * HDIM / 4);

    // node feature encoders
    email_embed_kernel<<<N_E / 16, 256, 0, stream>>>(email_x, Wh_email, b_email, e_buf);
    kan_tanh_kernel<8><<<N_U, HDIM, 0, stream>>>(url_x, url_bw, url_sw, u_buf);
    kan_tanh_kernel<1><<<N_S, HDIM, 0, stream>>>(sender_x, snd_bw, snd_sw, s_buf);

    // mean aggregation: sender->email and url->email
    count_kernel<<<(E_SE + 255) / 256, 256, 0, stream>>>(se_dst, cnt_se, E_SE);
    count_kernel<<<(E_UE + 255) / 256, 256, 0, stream>>>(ue_dst, cnt_ue, E_UE);
    {
        long tot = (long)E_SE * HDIM;
        scatter_add_kernel<<<(unsigned)((tot + 255) / 256), 256, 0, stream>>>(s_buf, se_src, se_dst, acc_se, tot);
    }
    {
        long tot = (long)E_UE * HDIM;
        scatter_add_kernel<<<(unsigned)((tot + 255) / 256), 256, 0, stream>>>(u_buf, ue_src, ue_dst, acc_ue, tot);
    }
    {
        long tot = (long)N_E * HDIM;
        norm_mean_kernel<<<(unsigned)((tot + 255) / 256), 256, 0, stream>>>(acc_se, cnt_se, tot);
        norm_mean_kernel<<<(unsigned)((tot + 255) / 256), 256, 0, stream>>>(acc_ue, cnt_ue, tot);
    }

    // fused SAGE linear layers -> email_h
    email_h_kernel<<<N_E / 16, 256, 0, stream>>>(acc_se, acc_ue, e_buf,
                                                 Wh_l_se, Wh_l_ue, Wh_sum, bl_se, bl_ue, h_buf);

    // batch-norm stats over concat([email_h, e])
    bn_partial_kernel<<<240, 256, 0, stream>>>(h_buf, e_buf, bn_sums, bn_sqs, N_E);
    bn_finalize_kernel<<<1, 256, 0, stream>>>(bn_sums, bn_sqs, stats, N_E);

    // fused BN + KAN classifier -> (N_E, 2)
    classify_kernel<<<N_E, 2 * HDIM, 0, stream>>>(h_buf, e_buf, stats, bn_gamma, bn_beta,
                                                  cls_bw, cls_sw, (float*)d_out);
}
